// EncoderLayer_83906481094755
// MI455X (gfx1250) — matrix operand order repaired and checked by reading, not compiled, not run on hardware
//
#include <hip/hip_runtime.h>

typedef __attribute__((ext_vector_type(16))) __bf16 v16bf;
typedef __attribute__((ext_vector_type(8)))  __bf16 v8bf;
typedef __attribute__((ext_vector_type(8)))  float  v8f;

namespace cfg {
constexpr int D  = 1024;   // model dim
constexpr int H  = 16;     // heads
constexpr int DH = 64;     // head dim
constexpr int F  = 4096;   // ffn hidden
constexpr int B  = 8;      // batch
constexpr int S  = 1024;   // seq
constexpr int R  = B * S;  // token rows = 8192
}

// ---- xor-butterfly within each 16-lane row: v_permlane16_b32, no LDS round-trip ----
template<int M>
__device__ __forceinline__ float rowxor(float x) {
#if __has_builtin(__builtin_amdgcn_permlane16)
  constexpr unsigned long long sel = []() {
    unsigned long long s = 0;
    for (int i = 0; i < 16; ++i) s |= (unsigned long long)((i ^ M) & 15) << (4 * i);
    return s;
  }();
  unsigned u = __builtin_bit_cast(unsigned, x);
  unsigned r = __builtin_amdgcn_permlane16(u, u, (unsigned)(sel & 0xffffffffULL),
                                           (unsigned)(sel >> 32), false, false);
  return __builtin_bit_cast(float, r);
#else
  return __shfl_xor(x, M, 32);
#endif
}

// ---- fragment loaders (CDNA5 wave32 WMMA bf16 16x16x32; contiguous k order, same in A and B) ----
// A (16x32 MxK): lane l holds row l%16; 16 contiguous elems K = k0..k0+15, k0 = (l<16?0:16)
__device__ __forceinline__ v16bf ldA(const __bf16* rowp, int k0l) {
  v8bf lo = *(const v8bf*)(rowp + k0l);
  v8bf hi = *(const v8bf*)(rowp + k0l + 8);
  return __builtin_shufflevector(lo, hi, 0,1,2,3,4,5,6,7,8,9,10,11,12,13,14,15);
}
// B (32x16 KxN): lane l holds col l%16; 16 contiguous K at offset (l<16?0:16) -> single 32B load

__device__ __forceinline__ v8f wmma_bf16(v16bf a, v16bf b, v8f c) {
  return __builtin_amdgcn_wmma_f32_16x16x32_bf16(false, a, false, b, (short)0, c, false, false);
}

// ---------------------------------------------------------------- cast f32->bf16 (8/thread)
__global__ void cast_f32_to_bf16(const float* __restrict__ in, __bf16* __restrict__ out, int n8) {
  int i = blockIdx.x * blockDim.x + threadIdx.x;
  if (i < n8) {
    v8f v = *(const v8f*)(in + (size_t)i * 8);
    v8bf o;
#pragma unroll
    for (int j = 0; j < 8; ++j) o[j] = (__bf16)v[j];
    *(v8bf*)(out + (size_t)i * 8) = o;
  }
}

// ---------------------------------------------------------------- GEMM: C = A * W^T + bias
// A [M,K] bf16 row-major; W [N,K] bf16 row-major (K contiguous on both).
// One wave -> 64x64 output tile, 4x4 f32 accumulators, K-loop step 32.
template<bool RELU, bool RESID, bool OUT_BF16, bool OUT_VT>
__global__ void gemm_bf16_nt(const __bf16* __restrict__ A, const __bf16* __restrict__ W,
                             const float* __restrict__ bias, const float* __restrict__ resid,
                             float* __restrict__ Cf, __bf16* __restrict__ Cb,
                             int M, int N, int K) {
  const int wave = (int)((blockIdx.x * blockDim.x + threadIdx.x) >> 5);
  const int lane = threadIdx.x & 31;
  const int tilesN = N >> 6;
  const int wm = wave / tilesN;
  const int wn = wave - wm * tilesN;
  if ((wm << 6) >= M) return;                // uniform per wave: EXEC stays all-1s for WMMA
  const int m0 = wm << 6, n0 = wn << 6;
  const int r   = lane & 15;
  const int k0a = (lane < 16) ? 0 : 16;
  const int k0b = (lane < 16) ? 0 : 16;

  v8f acc[4][4];
  {
    v8f z{};
#pragma unroll
    for (int i = 0; i < 4; ++i)
#pragma unroll
      for (int j = 0; j < 4; ++j) acc[i][j] = z;
  }

#pragma unroll 2
  for (int k = 0; k < K; k += 32) {
    v16bf a[4], bfr[4];
#pragma unroll
    for (int i = 0; i < 4; ++i)
      a[i] = ldA(A + (size_t)(m0 + i * 16 + r) * K + k, k0a);
#pragma unroll
    for (int j = 0; j < 4; ++j)
      bfr[j] = *(const v16bf*)(W + (size_t)(n0 + j * 16 + r) * K + k + k0b);
#pragma unroll
    for (int i = 0; i < 4; ++i)
#pragma unroll
      for (int j = 0; j < 4; ++j)
        acc[i][j] = wmma_bf16(a[i], bfr[j], acc[i][j]);
  }

  // Epilogue. C/D layout: VGPR e holds row e (lanes 0-15) / e+8 (lanes 16-31), col = lane%16.
#pragma unroll
  for (int j = 0; j < 4; ++j) {
    const int col = n0 + j * 16 + r;
    const float bv = bias[col];
#pragma unroll
    for (int i = 0; i < 4; ++i) {
      const int rbase = m0 + i * 16 + ((lane < 16) ? 0 : 8);
#pragma unroll
      for (int e = 0; e < 8; ++e) {
        const int row = rbase + e;
        float v = acc[i][j][e] + bv;
        if (RELU)  v = fmaxf(v, 0.0f);
        if (RESID) v += resid[(size_t)row * N + col];
        if (OUT_VT) {            // scatter V as V^T: [B,H,dh,S] so ctx-GEMM B-frags are contiguous
          const int bb = row / cfg::S, q = row % cfg::S;
          const int hh = col / cfg::DH, d = col % cfg::DH;
          Cb[(size_t)((bb * cfg::H + hh) * cfg::DH + d) * cfg::S + q] = (__bf16)v;
        } else if (OUT_BF16) {
          Cb[(size_t)row * N + col] = (__bf16)v;
        } else {
          Cf[(size_t)row * N + col] = v;
        }
      }
    }
  }
}

// ---------------------------------------------------------------- flash attention
// One wave per (b, h, 16-query tile). Streams 32-wide key blocks; scores via WMMA,
// online softmax with permlane16 butterflies, P relayout via per-wave LDS slice
// (DS ops are in-order within a wave -> no block barrier needed), ctx via WMMA.
__global__ void flash_attn(const __bf16* __restrict__ Q, const __bf16* __restrict__ Kc,
                           const __bf16* __restrict__ Vt, __bf16* __restrict__ ctx) {
  __shared__ __attribute__((aligned(32))) __bf16 lds[8][16 * 32];  // 1KB per wave
  const int wib  = threadIdx.x >> 5;
  const int lane = threadIdx.x & 31;
  const int wave = blockIdx.x * 8 + wib;
  const int r = lane & 15;
  const bool lo = lane < 16;
  const int k0a = lo ? 0 : 16;
  const int k0b = lo ? 0 : 16;
  const int qb = wave & (cfg::S / 16 - 1);
  const int h  = (wave >> 6) & (cfg::H - 1);
  const int b  = wave >> 10;
  const int q0 = qb << 4;

  const __bf16* qrow = Q + (size_t)(b * cfg::S + q0 + r) * cfg::D + h * cfg::DH;
  v16bf aQ[2];
#pragma unroll
  for (int c = 0; c < 2; ++c) aQ[c] = ldA(qrow + 32 * c, k0a);

  float mrow[8], lrow[8];
  v8f o[4];
  {
    v8f z{};
#pragma unroll
    for (int n = 0; n < 4; ++n) o[n] = z;
#pragma unroll
    for (int e = 0; e < 8; ++e) { mrow[e] = -1e30f; lrow[e] = 0.0f; }
  }

  for (int kb = 0; kb < cfg::S; kb += 32) {
    v8f s0{}, s1{};
#pragma unroll
    for (int c = 0; c < 2; ++c) {
      v16bf b0 = *(const v16bf*)(Kc + (size_t)(b * cfg::S + kb +      r) * cfg::D + h * cfg::DH + 32 * c + k0b);
      v16bf b1 = *(const v16bf*)(Kc + (size_t)(b * cfg::S + kb + 16 + r) * cfg::D + h * cfg::DH + 32 * c + k0b);
      s0 = wmma_bf16(aQ[c], b0, s0);
      s1 = wmma_bf16(aQ[c], b1, s1);
    }

    float corr[8];
#pragma unroll
    for (int e = 0; e < 8; ++e) {
      float x0 = s0[e] * 0.125f;          // 1/sqrt(64)
      float x1 = s1[e] * 0.125f;
      float mx = fmaxf(x0, x1);
      mx = fmaxf(mx, rowxor<1>(mx));
      mx = fmaxf(mx, rowxor<2>(mx));
      mx = fmaxf(mx, rowxor<4>(mx));
      mx = fmaxf(mx, rowxor<8>(mx));
      const float mn = fmaxf(mrow[e], mx);
      const float e0 = __expf(x0 - mn);
      const float e1 = __expf(x1 - mn);
      float rs = e0 + e1;
      rs += rowxor<1>(rs);
      rs += rowxor<2>(rs);
      rs += rowxor<4>(rs);
      rs += rowxor<8>(rs);
      const float c0 = __expf(mrow[e] - mn);
      lrow[e] = lrow[e] * c0 + rs;
      mrow[e] = mn;
      corr[e] = c0;
      const int prow = e + (lo ? 0 : 8);  // C-layout -> LDS row-major 16x32 P tile
      lds[wib][prow * 32 + r]      = (__bf16)e0;
      lds[wib][prow * 32 + 16 + r] = (__bf16)e1;
    }
#pragma unroll
    for (int n = 0; n < 4; ++n)
#pragma unroll
      for (int e = 0; e < 8; ++e) o[n][e] *= corr[e];

    v16bf aP = ldA(&lds[wib][r * 32], k0a);  // reload P in A-fragment layout (in-order DS per wave)
#pragma unroll
    for (int n = 0; n < 4; ++n) {
      v16bf bv = *(const v16bf*)(Vt + (size_t)((b * cfg::H + h) * cfg::DH + n * 16 + r) * cfg::S + kb + k0b);
      o[n] = wmma_bf16(aP, bv, o[n]);
    }
  }

  float invl[8];
#pragma unroll
  for (int e = 0; e < 8; ++e) invl[e] = 1.0f / lrow[e];
#pragma unroll
  for (int n = 0; n < 4; ++n)
#pragma unroll
    for (int e = 0; e < 8; ++e) {
      const int row = q0 + e + (lo ? 0 : 8);
      const float v = o[n][e] * invl[e];
      ctx[(size_t)(b * cfg::S + row) * cfg::D + h * cfg::DH + n * 16 + r] = (__bf16)v;
    }
}

// ---------------------------------------------------------------- LayerNorm (ddof=1, no eps/affine)
__global__ void layernorm_rows(const float* __restrict__ X, float* __restrict__ Yf,
                               __bf16* __restrict__ Yb) {
  __shared__ float sbuf[16];
  const int row = blockIdx.x;
  const int tid = threadIdx.x;
  const float* x = X + (size_t)row * cfg::D;
  float v[4]; float s = 0.0f, s2 = 0.0f;
#pragma unroll
  for (int i = 0; i < 4; ++i) { v[i] = x[tid + 256 * i]; s += v[i]; s2 += v[i] * v[i]; }
  s += rowxor<1>(s);  s2 += rowxor<1>(s2);
  s += rowxor<2>(s);  s2 += rowxor<2>(s2);
  s += rowxor<4>(s);  s2 += rowxor<4>(s2);
  s += rowxor<8>(s);  s2 += rowxor<8>(s2);
  s += __shfl_xor(s, 16, 32);  s2 += __shfl_xor(s2, 16, 32);  // cross-row step
  const int wib = tid >> 5;
  if ((tid & 31) == 0) { sbuf[wib] = s; sbuf[8 + wib] = s2; }
  __syncthreads();
  float S = 0.0f, S2 = 0.0f;
#pragma unroll
  for (int w = 0; w < 8; ++w) { S += sbuf[w]; S2 += sbuf[8 + w]; }
  const float mean = S * (1.0f / cfg::D);
  const float var  = (S2 - (float)cfg::D * mean * mean) * (1.0f / (cfg::D - 1));
  const float inv  = rsqrtf(var);
#pragma unroll
  for (int i = 0; i < 4; ++i) {
    const float y = (v[i] - mean) * inv;
    Yf[(size_t)row * cfg::D + tid + 256 * i] = y;
    if (Yb) Yb[(size_t)row * cfg::D + tid + 256 * i] = (__bf16)y;
  }
}

// ---------------------------------------------------------------- launcher
extern "C" void kernel_launch(void* const* d_in, const int* in_sizes, int n_in,
                              void* d_out, int out_size, void* d_ws, size_t ws_size,
                              hipStream_t stream) {
  using namespace cfg;
  (void)in_sizes; (void)n_in; (void)out_size; (void)ws_size;
  const float* x  = (const float*)d_in[0];
  const float* wq = (const float*)d_in[1];
  const float* bq = (const float*)d_in[2];
  const float* wk = (const float*)d_in[3];
  const float* bk = (const float*)d_in[4];
  const float* wv = (const float*)d_in[5];
  const float* bv = (const float*)d_in[6];
  const float* wo = (const float*)d_in[7];
  const float* bo = (const float*)d_in[8];
  const float* w1 = (const float*)d_in[9];
  const float* b1 = (const float*)d_in[10];
  const float* w2 = (const float*)d_in[11];
  const float* b2 = (const float*)d_in[12];
  float* out = (float*)d_out;

  char* ws = (char*)d_ws;
  size_t off = 0;
  auto alloc = [&](size_t bytes) { void* p = ws + off; off = (off + bytes + 255) & ~(size_t)255; return p; };
  __bf16* xb   = (__bf16*)alloc((size_t)R * D * 2);
  __bf16* wqb  = (__bf16*)alloc((size_t)D * D * 2);
  __bf16* wkb  = (__bf16*)alloc((size_t)D * D * 2);
  __bf16* wvb  = (__bf16*)alloc((size_t)D * D * 2);
  __bf16* wob  = (__bf16*)alloc((size_t)D * D * 2);
  __bf16* w1b  = (__bf16*)alloc((size_t)F * D * 2);
  __bf16* w2b  = (__bf16*)alloc((size_t)D * F * 2);
  __bf16* Qb   = (__bf16*)alloc((size_t)R * D * 2);
  __bf16* Kb   = (__bf16*)alloc((size_t)R * D * 2);
  __bf16* Vt   = (__bf16*)alloc((size_t)R * D * 2);   // [B,H,dh,S]
  __bf16* ctxb = (__bf16*)alloc((size_t)R * D * 2);
  float*  t1   = (float*) alloc((size_t)R * D * 4);
  float*  hf   = (float*) alloc((size_t)R * D * 4);
  __bf16* hb   = (__bf16*)alloc((size_t)R * D * 2);
  __bf16* ub   = (__bf16*)alloc((size_t)R * F * 2);
  float*  t2   = (float*) alloc((size_t)R * D * 4);

  auto cast = [&](const float* src, __bf16* dst, size_t n) {
    size_t n8 = n / 8;
    cast_f32_to_bf16<<<(unsigned)((n8 + 255) / 256), 256, 0, stream>>>(src, dst, (int)n8);
  };
  cast(x,  xb,  (size_t)R * D);
  cast(wq, wqb, (size_t)D * D);
  cast(wk, wkb, (size_t)D * D);
  cast(wv, wvb, (size_t)D * D);
  cast(wo, wob, (size_t)D * D);
  cast(w1, w1b, (size_t)F * D);
  cast(w2, w2b, (size_t)D * F);

  const dim3 blk(256);  // 8 waves/block
  const unsigned gD = (unsigned)(((R / 64) * (D / 64) + 7) / 8);  // M=8192,N=1024 -> 256 blocks
  const unsigned gF = (unsigned)(((R / 64) * (F / 64) + 7) / 8);  // M=8192,N=4096 -> 1024 blocks

  // QKV projections (V stored transposed per-head)
  gemm_bf16_nt<false,false,true ,false><<<gD, blk, 0, stream>>>(xb, wqb, bq, nullptr, nullptr, Qb, R, D, D);
  gemm_bf16_nt<false,false,true ,false><<<gD, blk, 0, stream>>>(xb, wkb, bk, nullptr, nullptr, Kb, R, D, D);
  gemm_bf16_nt<false,false,true ,true ><<<gD, blk, 0, stream>>>(xb, wvb, bv, nullptr, nullptr, Vt, R, D, D);

  // flash attention: B*H*(S/16) = 8192 waves -> 1024 blocks
  flash_attn<<<(unsigned)((B * H * (S / 16)) / 8), blk, 0, stream>>>(Qb, Kb, Vt, ctxb);

  // output projection + residual(x) -> t1 ; LN -> h (fp32 + bf16)
  gemm_bf16_nt<false,true ,false,false><<<gD, blk, 0, stream>>>(ctxb, wob, bo, x, t1, nullptr, R, D, D);
  layernorm_rows<<<R, blk, 0, stream>>>(t1, hf, hb);

  // FFN1 (+ReLU, bf16) ; FFN2 (+residual h) ; LN -> out
  gemm_bf16_nt<true ,false,true ,false><<<gF, blk, 0, stream>>>(hb, w1b, b1, nullptr, nullptr, ub, R, F, D);
  gemm_bf16_nt<false,true ,false,false><<<gD, blk, 0, stream>>>(ub, w2b, b2, hf, t2, nullptr, R, D, F);
  layernorm_rows<<<R, blk, 0, stream>>>(t2, out, nullptr);
}
